// MultiscaleTensorField_37056977830560
// MI455X (gfx1250) — compile-verified, matrix operand order checked by reading, not run on hardware
//
#include <hip/hip_runtime.h>
#include <hip/hip_bf16.h>
#include <stdint.h>

// Problem constants (from reference)
#define Q_      4096
#define N_      16384
#define K_      32
#define C_      128
#define H_      4
#define DH_     32
#define LEMB_   32
#define CTX_    64
#define SHN_    9
#define LAYERS_ 2
#define MH_     384
#define E_      (2 * Q_ * K_)     // 262144
#define QK_     (Q_ * K_)         // 131072
#define ESD_    (LEMB_ + CTX_)    // 96

#define LDS_STRIDE 40             // ushorts per row; 80B rows keep 16B alignment

typedef __attribute__((ext_vector_type(16))) __bf16          v16bf;
typedef __attribute__((ext_vector_type(8)))  float           v8f;
typedef __attribute__((ext_vector_type(16))) unsigned short  v16us;
typedef __attribute__((ext_vector_type(4)))  unsigned int    v4u;
typedef __attribute__((ext_vector_type(4)))  int             v4i;
typedef __attribute__((ext_vector_type(8)))  int             v8i;

union FragU { v16us u; v16bf b; };

__device__ __forceinline__ unsigned short f2bf(float f) {
  // round-to-nearest-even f32 -> bf16
  unsigned u = __builtin_bit_cast(unsigned, f);
  unsigned r = u + 0x7FFFu + ((u >> 16) & 1u);
  return (unsigned short)(r >> 16);
}

__device__ __forceinline__ float gelu_tanh(float x) {
  const float c = 0.7978845608028654f;  // sqrt(2/pi)
  float t = tanhf(c * (x + 0.044715f * x * x * x));
  return 0.5f * x * (1.0f + t);
}

// ---------------------------------------------------------------------------
// Tensor Data Mover: load a 64(row) x 32(col) bf16 tile from a row-major
// [rows x Kd] bf16 matrix into LDS at ldsOff, padding each 64B row out to
// 80B (LDS_STRIDE=40 ushorts).  OOB rows (>= rowsRemain) read as zero.
// D# fields per CDNA5 ISA.
// ---------------------------------------------------------------------------
__device__ __forceinline__ void tdm_load_tile_bf16(
    unsigned ldsOff, const unsigned short* gbase, unsigned rowStart, int Kd,
    int k0, unsigned rowsRemain) {
  unsigned long long gaddr =
      (unsigned long long)(uintptr_t)gbase +
      ((unsigned long long)rowStart * (unsigned)Kd + (unsigned)k0) * 2ull;
  v4u g0;
  g0.x = 1u;                                    // count=1, user descriptor
  g0.y = ldsOff;                                // lds_addr
  g0.z = (unsigned)gaddr;                       // global_addr[31:0]
  g0.w = ((unsigned)(gaddr >> 32) & 0x1FFFFFFu) // global_addr[56:32]
         | (2u << 30);                          // type=2 ("image")
  unsigned td0 = (unsigned)(Kd - k0);           // remaining cols from tile org
  unsigned td1 = rowsRemain;                    // remaining rows (clamps OOB)
  v8i g1;
  g1[0] = (int)((1u << 16)      // data_size = 2 bytes (bf16)
                | (1u << 20)    // pad_enable
                | (3u << 22)    // pad_interval: 16 dwords (one 64B tile row)
                | (3u << 25));  // pad_amount:   4 dwords (pad row to 80B)
  g1[1] = (int)((td0 & 0xFFFFu) << 16);                         // tensor_dim0 lo
  g1[2] = (int)(((td0 >> 16) & 0xFFFFu) | ((td1 & 0xFFFFu) << 16));
  g1[3] = (int)(((td1 >> 16) & 0xFFFFu) | (32u << 16));         // tile_dim0=32
  g1[4] = (int)64u;                                             // tile_dim1=64
  g1[5] = (int)(unsigned)Kd;                                    // dim0_stride lo
  g1[6] = 0;
  g1[7] = 0;
  v4i z4 = {0, 0, 0, 0};
#if defined(__clang_major__) && __clang_major__ >= 23
  v8i z8 = {0, 0, 0, 0, 0, 0, 0, 0};
  __builtin_amdgcn_tensor_load_to_lds(g0, g1, z4, z4, z8, 0);
#else
  __builtin_amdgcn_tensor_load_to_lds(g0, g1, z4, z4, 0);
#endif
}

// ---------------------------------------------------------------------------
// Kernel 0: one-off weight transpose+convert: Bt[n][k] = bf16(B[k][n])
// ---------------------------------------------------------------------------
__global__ void transpose_bf16_kernel(const float* __restrict__ in,
                                      unsigned short* __restrict__ out,
                                      int K, int N) {
  int idx = blockIdx.x * blockDim.x + threadIdx.x;
  if (idx >= K * N) return;
  int k = idx / N, n = idx - k * N;
  out[(size_t)n * K + k] = f2bf(in[idx]);
}

// ---------------------------------------------------------------------------
// Kernel 1: per-edge geometry -> sh9 (f32 E x 9) and edge_scalars (bf16 E x 96)
// ---------------------------------------------------------------------------
__global__ void edge_geom_kernel(const float* __restrict__ qpos,
                                 const float* __restrict__ spos,
                                 const float* __restrict__ ctx,
                                 const int* __restrict__ esrc,
                                 const int* __restrict__ edst,
                                 const int* __restrict__ escale,
                                 unsigned short* __restrict__ esc,
                                 float* __restrict__ sh9) {
  int e = blockIdx.x * blockDim.x + threadIdx.x;
  if (e >= E_) return;
  int d = edst[e], s = esrc[e], sc = escale[e];
  float cut = (sc == 0) ? 0.1f : 0.2f;
  float rx = qpos[d * 3 + 0] - spos[s * 3 + 0];
  float ry = qpos[d * 3 + 1] - spos[s * 3 + 1];
  float rz = qpos[d * 3 + 2] - spos[s * 3 + 2];
  float r = sqrtf(rx * rx + ry * ry + rz * rz + 1e-12f);
  float inv = 1.0f / r;
  float ux = rx * inv, uy = ry * inv, uz = rz * inv;
  float* sp = sh9 + (size_t)e * SHN_;
  sp[0] = 1.0f;
  sp[1] = ux; sp[2] = uy; sp[3] = uz;
  sp[4] = ux * uy; sp[5] = uy * uz;
  sp[6] = uz * uz - (1.0f / 3.0f);
  sp[7] = ux * uz; sp[8] = ux * ux - uy * uy;
  unsigned short* ep = esc + (size_t)e * ESD_;
  float coef = sqrtf(2.0f / cut) * inv;
  float arg = 3.14159265358979323846f * r / cut;
#pragma unroll
  for (int k = 1; k <= LEMB_; ++k)
    ep[k - 1] = f2bf(coef * __sinf((float)k * arg));
  const float* cp = ctx + (size_t)d * CTX_;
#pragma unroll 8
  for (int j = 0; j < CTX_; ++j) ep[LEMB_ + j] = f2bf(cp[j]);
}

// ---------------------------------------------------------------------------
// Kernel 2: bf16 WMMA GEMM, TDM/LDS staged.
//   C[M,N] = act(A[M,K] @ Bt[N,K]^T + bias) (+ residual)
//   Block tile 64(M) x 64(N); 8 waves as 4x2; each wave: 1 A-frag, 2 B-frags,
//   2 accumulators -> 2 v_wmma per K-step.
//   B tiles (and A tiles when A is bf16) are fetched by tensor_load_to_lds.
//   SHDOT epilogue: wdot[row] = sum_j C[row,j]*sh9[row,j]  (requires N<=16)
// Requirements: M % 64 == 0, K % 32 == 0.
// ---------------------------------------------------------------------------
template <bool ABF16, bool OUTBF16, bool GELU, bool SHDOT>
__global__ void wmma_gemm_kernel(const void* __restrict__ Aptr,
                                 const unsigned short* __restrict__ Bt,
                                 const float* __restrict__ bias,
                                 const float* __restrict__ residual,
                                 void* __restrict__ Cptr,
                                 const float* __restrict__ sh9,
                                 float* __restrict__ wdot,
                                 int M, int N, int Kd) {
  __shared__ __align__(16) unsigned short ldsA[64 * LDS_STRIDE];
  __shared__ __align__(16) unsigned short ldsB[64 * LDS_STRIDE];

  const int tid  = threadIdx.x;
  const int lane = tid & 31;
  const int wave = tid >> 5;
  const int tm   = wave & 3;   // 4 M sub-tiles
  const int tn   = wave >> 2;  // 2 N sub-tile pairs (each 32 wide)
  const int bm   = blockIdx.y * 64;
  const int bn   = blockIdx.x * 64;

  const float*          Af = (const float*)Aptr;
  const unsigned short* Ab = (const unsigned short*)Aptr;

  v8f acc0 = {0.f, 0.f, 0.f, 0.f, 0.f, 0.f, 0.f, 0.f};
  v8f acc1 = {0.f, 0.f, 0.f, 0.f, 0.f, 0.f, 0.f, 0.f};

  const unsigned ldsAoff = (unsigned)(uintptr_t)&ldsA[0];
  const unsigned ldsBoff = (unsigned)(uintptr_t)&ldsB[0];

  for (int k0 = 0; k0 < Kd; k0 += 32) {
    // wave 0 drives the Tensor Data Mover for the B tile (and A when bf16)
    if (wave == 0) {
      tdm_load_tile_bf16(ldsBoff, Bt, (unsigned)bn, Kd, k0,
                         (unsigned)(N - bn));
      if (ABF16)
        tdm_load_tile_bf16(ldsAoff, Ab, (unsigned)bm, Kd, k0, 64u);
    }
    if (!ABF16) {
      // f32 A: vectorized load+convert. thread -> row=tid>>2, colblk=(tid&3)*8
      int row = tid >> 2, cb = (tid & 3) * 8;
      const float4* src =
          (const float4*)&Af[(size_t)(bm + row) * Kd + k0 + cb];
      float4 f0 = src[0], f1 = src[1];
      uint4 packed;
      packed.x = (unsigned)f2bf(f0.x) | ((unsigned)f2bf(f0.y) << 16);
      packed.y = (unsigned)f2bf(f0.z) | ((unsigned)f2bf(f0.w) << 16);
      packed.z = (unsigned)f2bf(f1.x) | ((unsigned)f2bf(f1.y) << 16);
      packed.w = (unsigned)f2bf(f1.z) | ((unsigned)f2bf(f1.w) << 16);
      *(uint4*)&ldsA[row * LDS_STRIDE + cb] = packed;
      if (k0 + 32 < Kd)  // CDNA5 global_prefetch_b8 of next A tile
        __builtin_prefetch(&Af[(size_t)(bm + row) * Kd + k0 + 32 + cb], 0, 0);
    }
    if (wave == 0) __builtin_amdgcn_s_wait_tensorcnt(0);
    __syncthreads();

    // Fragments per the CDNA5 16-bit 16x32 layout:
    //   lanes 0-15: row=lane,    K = {0..7, 16..23}
    //   lanes16-31: row=lane-16, K = {8..15, 24..31};  VGPR p = K-pair (k,k+1)
    const int lm   = lane & 15;
    const int koff = (lane < 16) ? 0 : 8;
    FragU fa, fb0, fb1;
#pragma unroll
    for (int p = 0; p < 8; ++p) {
      int kk = (p < 4) ? (koff + 2 * p) : (16 + koff + 2 * (p - 4));
      const unsigned short* sa = &ldsA[(tm * 16 + lm) * LDS_STRIDE + kk];
      fa.u[2 * p] = sa[0];  fa.u[2 * p + 1] = sa[1];
      const unsigned short* sb0 = &ldsB[(tn * 32 + lm) * LDS_STRIDE + kk];
      fb0.u[2 * p] = sb0[0]; fb0.u[2 * p + 1] = sb0[1];
      const unsigned short* sb1 = &ldsB[(tn * 32 + 16 + lm) * LDS_STRIDE + kk];
      fb1.u[2 * p] = sb1[0]; fb1.u[2 * p + 1] = sb1[1];
    }
    acc0 = __builtin_amdgcn_wmma_f32_16x16x32_bf16(
        false, fa.b, false, fb0.b, (short)0, acc0, false, false);
    acc1 = __builtin_amdgcn_wmma_f32_16x16x32_bf16(
        false, fa.b, false, fb1.b, (short)0, acc1, false, false);
    __syncthreads();
  }

  // ----- Epilogue -----
  // C 16x16 f32 layout: lane L holds col (L&15); VGPR j = row j (+8 if L>=16)
#pragma unroll
  for (int s = 0; s < 2; ++s) {
    v8f acc = s ? acc1 : acc0;
    int nbase = bn + tn * 32 + s * 16;
    if (SHDOT) {
      if (nbase < N) {  // SHDOT used with N<=16: exactly one live sub-tile
        int col  = lane & 15;
        int gcol = nbase + col;
        bool cok = gcol < N;
        float bb = (cok && bias) ? bias[gcol] : 0.f;
#pragma unroll
        for (int j = 0; j < 8; ++j) {
          int row = bm + tm * 16 + ((lane < 16) ? 0 : 8) + j;
          float v = acc[j] + bb;
          float t = cok ? v * sh9[(size_t)row * SHN_ + gcol] : 0.f;
          t += __shfl_xor(t, 1, 32);
          t += __shfl_xor(t, 2, 32);
          t += __shfl_xor(t, 4, 32);
          t += __shfl_xor(t, 8, 32);
          if (col == 0) wdot[row] = t;
        }
      }
    } else {
      int col = nbase + (lane & 15);
      if (col < N) {
        float bb = bias ? bias[col] : 0.f;
#pragma unroll
        for (int j = 0; j < 8; ++j) {
          int row = bm + tm * 16 + ((lane < 16) ? 0 : 8) + j;
          float v = acc[j] + bb;
          if (GELU) v = gelu_tanh(v);
          if (residual) v += residual[(size_t)row * N + col];
          if (OUTBF16)
            ((unsigned short*)Cptr)[(size_t)row * N + col] = f2bf(v);
          else
            ((float*)Cptr)[(size_t)row * N + col] = v;
        }
      }
    }
  }
}

// ---------------------------------------------------------------------------
// Kernel 3: segmented softmax attention. One wave32 per (query, head).
// Edge layout (from reference builder): query q owns edges
//   scale0: q*32 + k, scale1: 131072 + q*32 + k  (k = lane)
// ---------------------------------------------------------------------------
__global__ void attn_kernel(const int* __restrict__ esrc,
                            const unsigned char* __restrict__ emask,
                            const float* __restrict__ Qf,
                            const float* __restrict__ Vm,
                            const float* __restrict__ wdot,
                            float* __restrict__ agg) {
  int wid  = (blockIdx.x * blockDim.x + threadIdx.x) >> 5;
  int lane = threadIdx.x & 31;
  if (wid >= Q_ * H_) return;
  int q = wid >> 2;  // / H_
  int h = wid & 3;   // % H_
  int e0 = q * K_ + lane;
  int e1 = QK_ + q * K_ + lane;
  int s0 = esrc[e0], s1 = esrc[e1];
  float wd0 = wdot[e0], wd1 = wdot[e1];

  const float* qv = Qf + (size_t)q * C_ + h * DH_;
  const float* v0 = Vm + (size_t)s0 * C_ + h * DH_;
  const float* v1 = Vm + (size_t)s1 * C_ + h * DH_;
  float d0 = 0.f, d1 = 0.f;
#pragma unroll
  for (int i = 0; i < DH_; ++i) {
    float qi = qv[i];
    d0 += qi * v0[i];
    d1 += qi * v1[i];
  }
  const float rs = 0.17677669529663687f;  // 1/sqrt(DH)
  float l0 = d0 * wd0 * rs; l0 = (l0 > 0.f) ? l0 : 0.2f * l0;
  float l1 = d1 * wd1 * rs; l1 = (l1 > 0.f) ? l1 : 0.2f * l1;
  float mf0 = emask[e0] ? 1.f : 0.f;
  float mf1 = emask[e1] ? 1.f : 0.f;
  l0 = (mf0 > 0.f) ? l0 : -1e30f;
  l1 = (mf1 > 0.f) ? l1 : -1e30f;

  float mx = fmaxf(l0, l1);
#pragma unroll
  for (int m = 16; m >= 1; m >>= 1) mx = fmaxf(mx, __shfl_xor(mx, m, 32));
  float ex0 = mf0 * __expf(l0 - mx);
  float ex1 = mf1 * __expf(l1 - mx);
  float den = ex0 + ex1;
#pragma unroll
  for (int m = 16; m >= 1; m >>= 1) den += __shfl_xor(den, m, 32);
  float inv = 1.0f / (den + 1e-12f);
  float c0 = ex0 * inv * wd0;  // attn * wdot (m = v * wdot)
  float c1 = ex1 * inv * wd1;

  float acc = 0.f;
  const float* Vh = Vm + h * DH_ + lane;  // lane = dh channel
#pragma unroll 4
  for (int e = 0; e < 32; ++e) {
    float c = __shfl(c0, e, 32);
    int  si = __shfl(s0, e, 32);
    acc += c * Vh[(size_t)si * C_];
  }
#pragma unroll 4
  for (int e = 0; e < 32; ++e) {
    float c = __shfl(c1, e, 32);
    int  si = __shfl(s1, e, 32);
    acc += c * Vh[(size_t)si * C_];
  }
  agg[(size_t)q * C_ + h * DH_ + lane] = acc;
}

// ---------------------------------------------------------------------------
// Host orchestration
// ---------------------------------------------------------------------------
extern "C" void kernel_launch(void* const* d_in, const int* in_sizes, int n_in,
                              void* d_out, int out_size, void* d_ws,
                              size_t ws_size, hipStream_t stream) {
  const float* qpos  = (const float*)d_in[0];
  const float* qfeat = (const float*)d_in[1];
  const float* spos  = (const float*)d_in[2];
  const float* sfeat = (const float*)d_in[3];
  const float* ctx   = (const float*)d_in[4];
  const float* Wq    = (const float*)d_in[5];
  const float* Wv    = (const float*)d_in[6];
  const float* Wo    = (const float*)d_in[7];
  const float* bo    = (const float*)d_in[8];
  const float* Wr1   = (const float*)d_in[9];
  const float* br1   = (const float*)d_in[10];
  const float* Wr2   = (const float*)d_in[11];
  const float* br2   = (const float*)d_in[12];
  const float* Wf1   = (const float*)d_in[13];
  const float* bf1   = (const float*)d_in[14];
  const float* Wf2   = (const float*)d_in[15];
  const float* bf2   = (const float*)d_in[16];
  const int* esrc    = (const int*)d_in[17];
  const int* edst    = (const int*)d_in[18];
  const int* escale  = (const int*)d_in[19];
  const unsigned char* emask = (const unsigned char*)d_in[20];  // jax bool = 1B

  char* ws = (char*)d_ws;
  size_t off = 0;
  auto take = [&](size_t bytes) -> char* {
    char* p = ws + off;
    off += (bytes + 255) & ~(size_t)255;
    return p;
  };
  unsigned short* esc = (unsigned short*)take((size_t)E_ * ESD_ * 2);
  float* sh9   = (float*)take((size_t)E_ * SHN_ * 4);
  float* wdot  = (float*)take((size_t)E_ * 4);
  unsigned short* H1 = (unsigned short*)take((size_t)E_ * 64 * 2);
  float* Vm    = (float*)take((size_t)2 * N_ * C_ * 4);
  float* Qm    = (float*)take((size_t)Q_ * C_ * 4);
  float* agg   = (float*)take((size_t)Q_ * C_ * 4);
  float* outb  = (float*)take((size_t)Q_ * C_ * 4);
  float* ffa   = (float*)take((size_t)Q_ * MH_ * 4);
  float* featA = (float*)take((size_t)Q_ * C_ * 4);
  // transposed bf16 weights, per layer
  unsigned short* wqt[LAYERS_];  unsigned short* wvt[LAYERS_];
  unsigned short* wot[LAYERS_];  unsigned short* wr1t[LAYERS_];
  unsigned short* wr2t[LAYERS_]; unsigned short* wf1t[LAYERS_];
  unsigned short* wf2t[LAYERS_];
  for (int l = 0; l < LAYERS_; ++l) {
    wqt[l]  = (unsigned short*)take((size_t)C_ * C_ * 2);
    wvt[l]  = (unsigned short*)take((size_t)C_ * C_ * 2);
    wot[l]  = (unsigned short*)take((size_t)C_ * C_ * 2);
    wr1t[l] = (unsigned short*)take((size_t)ESD_ * 64 * 2);
    wr2t[l] = (unsigned short*)take((size_t)64 * SHN_ * 2);
    wf1t[l] = (unsigned short*)take((size_t)C_ * MH_ * 2);
    wf2t[l] = (unsigned short*)take((size_t)MH_ * C_ * 2);
  }

  auto xpose = [&](const float* w, unsigned short* t, int K, int N) {
    int tot = K * N;
    transpose_bf16_kernel<<<(tot + 255) / 256, 256, 0, stream>>>(w, t, K, N);
  };
  for (int l = 0; l < LAYERS_; ++l) {
    xpose(Wq  + (size_t)l * C_ * C_,    wqt[l],  C_,   C_);
    xpose(Wv  + (size_t)l * C_ * C_,    wvt[l],  C_,   C_);
    xpose(Wo  + (size_t)l * C_ * C_,    wot[l],  C_,   C_);
    xpose(Wr1 + (size_t)l * ESD_ * 64,  wr1t[l], ESD_, 64);
    xpose(Wr2 + (size_t)l * 64 * SHN_,  wr2t[l], 64,   SHN_);
    xpose(Wf1 + (size_t)l * C_ * MH_,   wf1t[l], C_,   MH_);
    xpose(Wf2 + (size_t)l * MH_ * C_,   wf2t[l], MH_,  C_);
  }

  edge_geom_kernel<<<E_ / 256, 256, 0, stream>>>(qpos, spos, ctx, esrc, edst,
                                                 escale, esc, sh9);

  const float* feat = qfeat;
  for (int l = 0; l < LAYERS_; ++l) {
    const float* bol  = bo  + (size_t)l * C_;
    const float* br1l = br1 + (size_t)l * 64;
    const float* br2l = br2 + (size_t)l * SHN_;
    const float* bf1l = bf1 + (size_t)l * MH_;
    const float* bf2l = bf2 + (size_t)l * C_;

    // H1 = gelu(edge_scalars @ Wr1 + br1)  [E x 64], bf16 out, A bf16 via TDM
    {
      dim3 g(1, E_ / 64);
      wmma_gemm_kernel<true, true, true, false><<<g, 256, 0, stream>>>(
          esc, wr1t[l], br1l, nullptr, H1, nullptr, nullptr, E_, 64, ESD_);
    }
    // wdot[e] = sum_j (H1 @ Wr2 + br2)[e,j] * sh9[e,j]   (fused SH-dot)
    {
      dim3 g(1, E_ / 64);
      wmma_gemm_kernel<true, false, false, true><<<g, 256, 0, stream>>>(
          H1, wr2t[l], br2l, nullptr, nullptr, sh9, wdot, E_, SHN_, 64);
    }
    // V = src_feat @ Wv   [32768 x 128]
    {
      dim3 g(C_ / 64, (2 * N_) / 64);
      wmma_gemm_kernel<false, false, false, false><<<g, 256, 0, stream>>>(
          sfeat, wvt[l], nullptr, nullptr, Vm, nullptr, nullptr, 2 * N_, C_,
          C_);
    }
    // Qm = feat @ Wq   [4096 x 128]
    {
      dim3 g(C_ / 64, Q_ / 64);
      wmma_gemm_kernel<false, false, false, false><<<g, 256, 0, stream>>>(
          feat, wqt[l], nullptr, nullptr, Qm, nullptr, nullptr, Q_, C_, C_);
    }
    // segmented softmax attention -> agg [4096 x 128]
    attn_kernel<<<(Q_ * H_) / 8, 256, 0, stream>>>(esrc, emask, Qm, Vm, wdot,
                                                   agg);
    // out = agg @ Wo + bo + feat
    {
      dim3 g(C_ / 64, Q_ / 64);
      wmma_gemm_kernel<false, false, false, false><<<g, 256, 0, stream>>>(
          agg, wot[l], bol, feat, outb, nullptr, nullptr, Q_, C_, C_);
    }
    // ffa = gelu(out @ Wf1 + bf1)   [4096 x 384]
    {
      dim3 g(MH_ / 64, Q_ / 64);
      wmma_gemm_kernel<false, false, true, false><<<g, 256, 0, stream>>>(
          outb, wf1t[l], bf1l, nullptr, ffa, nullptr, nullptr, Q_, MH_, C_);
    }
    // feat_next = out + ffa @ Wf2 + bf2
    float* dst = (l == LAYERS_ - 1) ? (float*)d_out : featA;
    {
      dim3 g(C_ / 64, Q_ / 64);
      wmma_gemm_kernel<false, false, false, false><<<g, 256, 0, stream>>>(
          ffa, wf2t[l], bf2l, outb, dst, nullptr, nullptr, Q_, C_, MH_);
    }
    feat = featA;
  }
}
